// LayerEnsembleCorMLP_910533067016
// MI455X (gfx1250) — matrix operand order repaired and checked by reading, not compiled, not run on hardware
//
#include <hip/hip_runtime.h>

// ---------------------------------------------------------------------------
// LayerEnsembleCorMLP on gfx1250 (MI455X): 3 ensemble-selected GEMM layers.
//   S=64 samples, B=128 batch, dims 1024 -> 1024 -> 1024 -> 256, E=8 ensembles.
// Strategy: one-shot weight transpose+bf16 downconvert to (e, n, k) layout,
// then tiled bf16 WMMA GEMMs (v_wmma_f32_16x16x32_bf16) with f32 accumulate.
// Tile staging via the Tensor Data Mover (tensor_load_to_lds + TENSORcnt),
// double-buffered so DMA of chunk i+1 overlaps WMMA of chunk i.
// Workspace need: ~68.5 MB.
// ---------------------------------------------------------------------------

typedef __attribute__((ext_vector_type(16))) __bf16 v16bf;
typedef __attribute__((ext_vector_type(8)))  __bf16 v8bf;
typedef __attribute__((ext_vector_type(8)))  float  v8f;

#define S_SAMPLES 64
#define BATCH     128
#define E_ENS     8
#define KC        32   // K chunk per WMMA (bf16 K=32)
#define TN        64   // N tile per block

#if defined(__HIP_DEVICE_COMPILE__)
#  if __has_builtin(__builtin_amdgcn_tensor_load_to_lds)
#    define USE_TDM 1
#  else
#    define USE_TDM 0
#  endif
#else
#  define USE_TDM 0
#endif

// ---- prep: f32 -> bf16 elementwise (inputs) -------------------------------
__global__ void cvt_f32_bf16(const float* __restrict__ in,
                             unsigned short* __restrict__ out, int n) {
  int i = blockIdx.x * blockDim.x + threadIdx.x;
  if (i < n) ((__bf16*)out)[i] = (__bf16)in[i];
}

// ---- prep: weight transpose (K,N,E) f32 -> (E,N,K) bf16 -------------------
__global__ __launch_bounds__(256)
void wt_transpose(const float* __restrict__ w, unsigned short* __restrict__ wt,
                  int K, int N) {
  __shared__ float tile[64][33];  // pad to kill bank conflicts on phase 2
  const int tilesN = N / 4;
  const int k0 = (blockIdx.x / tilesN) * 64;
  const int n0 = (blockIdx.x % tilesN) * 4;
  const int t = threadIdx.x;

  for (int i = t; i < 64 * 32; i += 256) {
    int k = i >> 5, c = i & 31;  // c = n_local*8 + e, contiguous in memory
    tile[k][c] = w[((size_t)(k0 + k) * N + n0) * E_ENS + c];
  }
  __syncthreads();

  __bf16* o = (__bf16*)wt;
  for (int i = t; i < 32 * 64; i += 256) {
    int c = i >> 6, k = i & 63;
    int nl = c >> 3, e = c & 7;
    o[((size_t)e * N + (n0 + nl)) * K + k0 + k] = (__bf16)tile[k][c];
  }
}

// ---- TDM helper: 2D tile (rows x KC bf16, row stride K) -> contiguous LDS --
#if USE_TDM
typedef unsigned int tdm_v4u __attribute__((ext_vector_type(4)));
typedef int          tdm_v4i __attribute__((ext_vector_type(4)));
typedef int          tdm_v8i __attribute__((ext_vector_type(8)));

__device__ __forceinline__ void tdm_load_tile(unsigned lds_addr,
                                              const __bf16* g,
                                              int rows, int K) {
  unsigned long long ga = (unsigned long long)(size_t)g;
  // D# group 0: count=1 (valid, user mode), lds_addr, 57-bit global addr,
  // type=2 ("image") in bits [127:126].
  tdm_v4u g0;
  g0.x = 1u;
  g0.y = lds_addr;
  g0.z = (unsigned)ga;
  g0.w = (unsigned)((ga >> 32) & 0x01FFFFFFu) | (2u << 30);
  // D# group 1: workgroup_mask=0, data_size=1 (2 bytes), no pad/iterate,
  // tensor_dim0=K, tensor_dim1=rows, tile_dim0=KC, tile_dim1=rows,
  // tensor_dim0_stride=K (elements).
  tdm_v8i g1;
  g1[0] = 1 << 16;                                               // data_size=2B
  g1[1] = (int)(((unsigned)K & 0xFFFFu) << 16);                  // dim0 lo
  g1[2] = (int)(((unsigned)K >> 16) | (((unsigned)rows) << 16)); // dim0 hi|dim1 lo
  g1[3] = (int)((unsigned)KC << 16);                             // dim1 hi|tile0
  g1[4] = rows & 0xFFFF;                                         // tile1 (tile2=0)
  g1[5] = K;                                                     // stride0 lo
  g1[6] = 0;                                                     // stride0 hi|stride1 lo
  g1[7] = 0;
  tdm_v4i z4 = {0, 0, 0, 0};
#if __clang_major__ >= 23
  tdm_v8i z8 = {0, 0, 0, 0, 0, 0, 0, 0};
  __builtin_amdgcn_tensor_load_to_lds(g0, g1, z4, z4, z8, 0);
#else
  __builtin_amdgcn_tensor_load_to_lds(g0, g1, z4, z4, 0);
#endif
}
#endif  // USE_TDM

// ---- shared WMMA inner step ----------------------------------------------
__device__ __forceinline__ void wmma_step(const __bf16* lA, const __bf16* lB,
                                          int wave, int lane, v8f acc[4]) {
  // A fragment: row m = 16*wave + lane%16; lanes 0-15: K 0..15,
  // lanes 16-31: K 16..31 (contiguous, matching B).
  const int rm  = lane & 15;
  const int khA = (lane < 16) ? 0 : 16;
  const int khB = (lane < 16) ? 0 : 16;
  union { v16bf v; v8bf h[2]; } af;
  const __bf16* ap = lA + (wave * 16 + rm) * KC + khA;
  af.h[0] = *(const v8bf*)ap;
  af.h[1] = *(const v8bf*)(ap + 8);
#pragma unroll
  for (int j = 0; j < 4; ++j) {
    // B fragment: column n = 16*j + lane%16; lanes 0-15 hold K 0..15,
    // lanes 16-31 hold K 16..31 (contiguous 32B in n-major LDS).
    const v16bf bfrag = *(const v16bf*)(lB + (j * 16 + rm) * KC + khB);
    acc[j] = __builtin_amdgcn_wmma_f32_16x16x32_bf16(
        false, af.v, false, bfrag, (short)0, acc[j], false, false);
  }
}

// ---- main GEMM: out[s] = act(A[s] @ Wt[e(s)]^T + b) -----------------------
// Block: 256 threads (8 waves), computes a 128(M) x 64(N) tile for sample s.
__global__ __launch_bounds__(256)
void ensemble_gemm(const unsigned short* __restrict__ Araw,   // bf16 [S?][128][K]
                   const unsigned short* __restrict__ Wtraw,  // bf16 [E][N][K]
                   const float* __restrict__ bias,            // f32 [N][E]
                   const long long* __restrict__ samples,     // i64 [S][3]
                   int layer, int K, int N,
                   float* __restrict__ outF,                  // f32 [S][128][N] or null
                   unsigned short* __restrict__ outBraw,      // bf16 [S][128][N] or null
                   int relu, int aShared) {
  const __bf16* A  = (const __bf16*)Araw;
  const __bf16* Wt = (const __bf16*)Wtraw;

  const int tilesN = N / TN;
  const int s  = blockIdx.x / tilesN;
  const int n0 = (blockIdx.x % tilesN) * TN;
  const int e  = (int)samples[s * 3 + layer];

  const __bf16* Ab = A + (aShared ? (size_t)0 : (size_t)s * BATCH * K);
  const __bf16* Bb = Wt + ((size_t)e * N + n0) * K;

  const int t    = threadIdx.x;
  const int lane = t & 31;   // wave32
  const int wave = t >> 5;   // 0..7

  v8f acc[4] = {};

#if USE_TDM
  // Double-buffered TDM pipeline: wave 0 drives the Tensor Data Mover; DMA of
  // chunk i+1 overlaps WMMA compute of chunk i. One barrier per iteration.
  __shared__ alignas(32) __bf16 ldsA[2][BATCH * KC];  // 2 x 8 KB
  __shared__ alignas(32) __bf16 ldsB[2][TN * KC];     // 2 x 4 KB

  const int wuni = __builtin_amdgcn_readfirstlane(threadIdx.x) >> 5;
  const unsigned aAddr[2] = {(unsigned)(size_t)&ldsA[0][0],
                             (unsigned)(size_t)&ldsA[1][0]};
  const unsigned bAddr[2] = {(unsigned)(size_t)&ldsB[0][0],
                             (unsigned)(size_t)&ldsB[1][0]};
  const int nch = K / KC;

  if (wuni == 0) {
    tdm_load_tile(aAddr[0], Ab, BATCH, K);
    tdm_load_tile(bAddr[0], Bb, TN, K);
  }
  for (int i = 0; i < nch; ++i) {
    if (wuni == 0) __builtin_amdgcn_s_wait_tensorcnt(0);  // chunk i landed
    __syncthreads();  // publish chunk i; all done reading both buffers
    if (wuni == 0 && i + 1 < nch) {
      const int kn = (i + 1) * KC;
      tdm_load_tile(aAddr[(i + 1) & 1], Ab + kn, BATCH, K);
      tdm_load_tile(bAddr[(i + 1) & 1], Bb + kn, TN, K);
    }
    wmma_step(ldsA[i & 1], ldsB[i & 1], wave, lane, acc);
  }
#else
  // Fallback: per-thread b128 staging.
  __shared__ alignas(32) __bf16 ldsA1[BATCH * KC];
  __shared__ alignas(32) __bf16 ldsB1[TN * KC];

  for (int k0 = 0; k0 < K; k0 += KC) {
#pragma unroll
    for (int c = 0; c < 2; ++c) {
      int idx = t + c * 256;
      int row = idx >> 2, seg = idx & 3;
      *(uint4*)(ldsA1 + row * KC + seg * 8) =
          *(const uint4*)(Ab + (size_t)row * K + k0 + seg * 8);
    }
    {
      int row = t >> 2, seg = t & 3;
      *(uint4*)(ldsB1 + row * KC + seg * 8) =
          *(const uint4*)(Bb + (size_t)row * K + k0 + seg * 8);
    }
    if (k0 + KC < K) {
      __builtin_prefetch(Ab + (size_t)(t >> 1) * K + k0 + KC, 0, 1);
      __builtin_prefetch(Bb + (size_t)(t >> 2) * K + k0 + KC, 0, 1);
    }
    __syncthreads();
    wmma_step(ldsA1, ldsB1, wave, lane, acc);
    __syncthreads();
  }
#endif

  // Epilogue: C/D layout is VGPR r -> M = 16*wave + r (+8 for lanes 16..31),
  // N = lane%16 within each 16-col group.
  const int mbase = wave * 16 + ((lane < 16) ? 0 : 8);
#pragma unroll
  for (int j = 0; j < 4; ++j) {
    const int n = n0 + j * 16 + (lane & 15);
    const float bv = bias[(size_t)n * E_ENS + e];
#pragma unroll
    for (int r = 0; r < 8; ++r) {
      float v = acc[j][r] + bv;
      if (relu) v = fmaxf(v, 0.0f);
      const size_t o = ((size_t)s * BATCH + mbase + r) * (size_t)N + n;
      if (outF) outF[o] = v;
      else      ((__bf16*)outBraw)[o] = (__bf16)v;
    }
  }
}

// ---------------------------------------------------------------------------
extern "C" void kernel_launch(void* const* d_in, const int* in_sizes, int n_in,
                              void* d_out, int out_size, void* d_ws, size_t ws_size,
                              hipStream_t stream) {
  const float*     inputs  = (const float*)d_in[0];
  const float*     w0      = (const float*)d_in[1];
  const float*     b0      = (const float*)d_in[2];
  const float*     w1      = (const float*)d_in[3];
  const float*     b1      = (const float*)d_in[4];
  const float*     w2      = (const float*)d_in[5];
  const float*     b2      = (const float*)d_in[6];
  const long long* samples = (const long long*)d_in[7];
  (void)in_sizes; (void)n_in; (void)out_size; (void)ws_size;

  char* ws = (char*)d_ws;
  size_t off = 0;
  auto alloc = [&](size_t bytes) -> void* {
    void* p = ws + off;
    off = (off + bytes + 255) & ~(size_t)255;
    return p;
  };
  unsigned short* inB  = (unsigned short*)alloc((size_t)BATCH * 1024 * 2);
  unsigned short* Wt0  = (unsigned short*)alloc((size_t)E_ENS * 1024 * 1024 * 2);
  unsigned short* Wt1  = (unsigned short*)alloc((size_t)E_ENS * 1024 * 1024 * 2);
  unsigned short* Wt2  = (unsigned short*)alloc((size_t)E_ENS * 256  * 1024 * 2);
  unsigned short* act1 = (unsigned short*)alloc((size_t)S_SAMPLES * BATCH * 1024 * 2);
  unsigned short* act2 = (unsigned short*)alloc((size_t)S_SAMPLES * BATCH * 1024 * 2);

  cvt_f32_bf16<<<(BATCH * 1024 + 255) / 256, 256, 0, stream>>>(inputs, inB, BATCH * 1024);
  wt_transpose<<<(1024 / 64) * (1024 / 4), 256, 0, stream>>>(w0, Wt0, 1024, 1024);
  wt_transpose<<<(1024 / 64) * (1024 / 4), 256, 0, stream>>>(w1, Wt1, 1024, 1024);
  wt_transpose<<<(1024 / 64) * (256  / 4), 256, 0, stream>>>(w2, Wt2, 1024, 256);

  ensemble_gemm<<<S_SAMPLES * (1024 / TN), 256, 0, stream>>>(
      inB, Wt0, b0, samples, 0, 1024, 1024, nullptr, act1, 1, 1);
  ensemble_gemm<<<S_SAMPLES * (1024 / TN), 256, 0, stream>>>(
      act1, Wt1, b1, samples, 1, 1024, 1024, nullptr, act2, 1, 0);
  ensemble_gemm<<<S_SAMPLES * (256 / TN), 256, 0, stream>>>(
      act2, Wt2, b2, samples, 2, 1024, 256, (float*)d_out, nullptr, 0, 0);
}
